// AMIPRouterTrain_51522427683260
// MI455X (gfx1250) — matrix operand order repaired and checked by reading, not compiled, not run on hardware
//
#include <hip/hip_runtime.h>
#include <hip/hip_bf16.h>
#include <math.h>

// Problem sizes (fixed by the reference)
#define NROWS 8192
#define DDIM  4096
#define KEXP  8
#define FHID  1024
#define CDIM  8192  // 2*D

// Workspace layout: [0, 256KB) routing weights f32 [N,K];
//                   [256KB, 256KB + 128MB) h1 bf16 [K,N,F]
#define WS_WEIGHTS_BYTES (NROWS * KEXP * 4)

typedef __attribute__((ext_vector_type(16))) __bf16 bf16x16;
typedef __attribute__((ext_vector_type(8)))  float  f32x8;

union FragAB {
  bf16x16 v;
  uint4   q[2];
};

// Exact RNE (epilogue only; 3 VALU/elem)
__device__ __forceinline__ unsigned short f32_to_bf16_rne(float f) {
  union { float f; unsigned int u; } x;
  x.f = f;
  unsigned int r = x.u + 0x7FFFu + ((x.u >> 16) & 1u);
  return (unsigned short)(r >> 16);
}

// Hot-loop conversion: round-half-up + v_perm_b32 byte pack.
// result[15:0] = bf16(x), result[31:16] = bf16(y)   (1.5 VALU/elem)
__device__ __forceinline__ unsigned int pack2_bf16(float x, float y) {
  union { float f; unsigned int u; } a, b;
  a.f = x;
  b.f = y;
  const unsigned int au = a.u + 0x8000u;
  const unsigned int bu = b.u + 0x8000u;
  return __builtin_amdgcn_perm(au, bu, 0x03020706u);
}

// CDNA5 async global->LDS (16B per lane, tracked on ASYNCcnt).
// lds_byte_off is the byte address within the workgroup's (single) LDS block.
__device__ __forceinline__ void async_copy_b128(unsigned int lds_byte_off,
                                                const void* gaddr) {
  asm volatile("global_load_async_to_lds_b128 %0, %1, off"
               :
               : "v"(lds_byte_off), "v"((unsigned long long)(uintptr_t)gaddr)
               : "memory");
}
__device__ __forceinline__ void wait_async_zero() {
  asm volatile("s_wait_asynccnt 0" ::: "memory");
}

// ---- tile staging helpers: ROWS x 32 reduction-cols per stage --------------
// 256 threads; 8 float4 per 32-col row.

template <int ROWS>
__device__ __forceinline__ void tile_load_f32_T(const float* __restrict__ src,
                                                int ldm, int t, float4* r) {
  constexpr int ITER = ROWS / 32;
#pragma unroll
  for (int i = 0; i < ITER; ++i) {
    int linear = t + 256 * i;
    int row = linear >> 3;
    int c4  = linear & 7;
    r[i] = *(const float4*)(src + (size_t)row * ldm + c4 * 4);
  }
}

template <int ROWS>
__device__ __forceinline__ void tile_store_bf16_T(unsigned short* lds, int t,
                                                  const float4* r) {
  constexpr int ITER = ROWS / 32;
#pragma unroll
  for (int i = 0; i < ITER; ++i) {
    int linear = t + 256 * i;
    int row = linear >> 3;
    int c4  = linear & 7;
    uint2 p;
    p.x = pack2_bf16(r[i].x, r[i].y);
    p.y = pack2_bf16(r[i].z, r[i].w);
    *(uint2*)(lds + row * 32 + c4 * 4) = p;
  }
}

// A fragment (16x32 bf16): lane = M row; lanes<16 hold K {0..7,16..23},
// lanes>=16 hold K {8..15,24..31} (two contiguous 16B chunks).
__device__ __forceinline__ void load_fragA(const unsigned short* lds, int mbase,
                                           int lane, FragAB& f) {
  const int row = mbase + (lane & 15);
  const int ch  = (lane & 16) ? 8 : 0;
  f.q[0] = *(const uint4*)(lds + row * 32 + ch);
  f.q[1] = *(const uint4*)(lds + row * 32 + 16 + ch);
}

// B fragment (32x16 bf16): lane = N column; lanes<16 hold K {0..7,16..23},
// lanes>=16 hold K {8..15,24..31} (two contiguous 16B chunks).
__device__ __forceinline__ void load_fragB(const unsigned short* lds, int nbase,
                                           int lane, FragAB& f) {
  const int col = nbase + (lane & 15);
  const int k0  = (lane & 16) ? 8 : 0;
  f.q[0] = *(const uint4*)(lds + col * 32 + k0);
  f.q[1] = *(const uint4*)(lds + col * 32 + 16 + k0);
}

// ---------------------------------------------------------------------------
// Kernel 1: routing softmax  weights[n,k] = softmax_k(h_mask[n,:]·Wr[k,:]+br)
// ---------------------------------------------------------------------------
__global__ __launch_bounds__(256) void moe_routing(
    const float* __restrict__ h_mask, const float* __restrict__ Wr,
    const float* __restrict__ br, float* __restrict__ weights) {
  const int lane = threadIdx.x & 31;
  const int wave = threadIdx.x >> 5;
  const int n = blockIdx.x * 8 + wave;

  float s[KEXP];
#pragma unroll
  for (int k = 0; k < KEXP; ++k) s[k] = 0.f;

  const float* hm = h_mask + (size_t)n * DDIM;
  for (int c = lane * 4; c < DDIM; c += 32 * 4) {
    const float4 hv = *(const float4*)(hm + c);
#pragma unroll
    for (int k = 0; k < KEXP; ++k) {
      const float4 wv = *(const float4*)(Wr + (size_t)k * DDIM + c);
      s[k] += hv.x * wv.x + hv.y * wv.y + hv.z * wv.z + hv.w * wv.w;
    }
  }
#pragma unroll
  for (int k = 0; k < KEXP; ++k) {
#pragma unroll
    for (int off = 16; off >= 1; off >>= 1) s[k] += __shfl_xor(s[k], off, 32);
    s[k] += br[k];
  }
  float m = s[0];
#pragma unroll
  for (int k = 1; k < KEXP; ++k) m = fmaxf(m, s[k]);
  float sum = 0.f;
#pragma unroll
  for (int k = 0; k < KEXP; ++k) {
    s[k] = expf(s[k] - m);
    sum += s[k];
  }
  const float inv = 1.f / sum;
  if (lane == 0) {
#pragma unroll
    for (int k = 0; k < KEXP; ++k) weights[(size_t)n * KEXP + k] = s[k] * inv;
  }
}

// ---------------------------------------------------------------------------
// Kernel 2: h1[k,n,f] = bf16(gelu(conditioned[n,:] @ W1[k,f,:] + b1[k,f]))
// Grid (N/128, F/256, K); block tile 128(M) x 256(N), 8 waves as 2x4,
// wave tile 64x64 = 4x4 WMMA fragments -> 16 wmma per 32-K step.
// Single-barrier double-buffered pipeline.
// ---------------------------------------------------------------------------
__global__ __launch_bounds__(256) void moe_gemm1_gelu(
    const float* __restrict__ h_anchor, const float* __restrict__ h_mask,
    const float* __restrict__ W1, const float* __restrict__ b1,
    unsigned short* __restrict__ h1) {
  __shared__ unsigned short ldsA[2][128 * 32];
  __shared__ unsigned short ldsB[2][256 * 32];

  const int t = threadIdx.x;
  const int lane = t & 31;
  const int wave = t >> 5;
  const int waveM = wave >> 2;  // 0..1 -> rows 0/64
  const int waveN = wave & 3;   // 0..3 -> cols 0/64/128/192
  const int nTile = blockIdx.x * 128;
  const int fTile = blockIdx.y * 256;
  const int k = blockIdx.z;

  const float* __restrict__ W1k = W1 + (size_t)k * FHID * CDIM;

  f32x8 acc[4][4];
#pragma unroll
  for (int mi = 0; mi < 4; ++mi)
#pragma unroll
    for (int ni = 0; ni < 4; ++ni)
#pragma unroll
      for (int r = 0; r < 8; ++r) acc[mi][ni][r] = 0.f;

  const int KT = CDIM / 32;  // 256 reduction stages
  float4 ra[4], rb[8];

  // stage 0 (col base 0 < D -> always h_anchor)
  tile_load_f32_T<128>(h_anchor + (size_t)nTile * DDIM, DDIM, t, ra);
  tile_load_f32_T<256>(W1k + (size_t)fTile * CDIM, CDIM, t, rb);
  tile_store_bf16_T<128>(ldsA[0], t, ra);
  tile_store_bf16_T<256>(ldsB[0], t, rb);
  __syncthreads();

  for (int kk = 0; kk < KT; ++kk) {
    const int buf = kk & 1;
    if (kk + 1 < KT) {
      const int cc = (kk + 1) * 32;  // whole stage comes from one source
      const float* srcA = (cc < DDIM)
                              ? (h_anchor + (size_t)nTile * DDIM + cc)
                              : (h_mask + (size_t)nTile * DDIM + (cc - DDIM));
      tile_load_f32_T<128>(srcA, DDIM, t, ra);
      tile_load_f32_T<256>(W1k + (size_t)fTile * CDIM + cc, CDIM, t, rb);
      if (kk + 2 < KT)  // prefetch the weight stream two stages ahead
        __builtin_prefetch(W1k + (size_t)(fTile + t) * CDIM + (kk + 2) * 32, 0, 3);
    }

    FragAB a[4], b[4];
#pragma unroll
    for (int mi = 0; mi < 4; ++mi)
      load_fragA(ldsA[buf], waveM * 64 + mi * 16, lane, a[mi]);
#pragma unroll
    for (int ni = 0; ni < 4; ++ni)
      load_fragB(ldsB[buf], waveN * 64 + ni * 16, lane, b[ni]);
#pragma unroll
    for (int mi = 0; mi < 4; ++mi)
#pragma unroll
      for (int ni = 0; ni < 4; ++ni)
        acc[mi][ni] = __builtin_amdgcn_wmma_f32_16x16x32_bf16(
            false, a[mi].v, false, b[ni].v, (short)0, acc[mi][ni], false, false);

    // single publishing barrier: prior readers of buf^1 were ordered by the
    // previous stage's barrier, so stores can go straight in.
    if (kk + 1 < KT) {
      tile_store_bf16_T<128>(ldsA[buf ^ 1], t, ra);
      tile_store_bf16_T<256>(ldsB[buf ^ 1], t, rb);
      __syncthreads();
    }
  }

  // epilogue: +b1, exact-erf GELU, store bf16 h1[k][n][f]
#pragma unroll
  for (int ni = 0; ni < 4; ++ni) {
    const int f = fTile + waveN * 64 + ni * 16 + (lane & 15);
    const float b1v = b1[k * FHID + f];
#pragma unroll
    for (int mi = 0; mi < 4; ++mi) {
#pragma unroll
      for (int r = 0; r < 8; ++r) {
        const int n = nTile + waveM * 64 + mi * 16 + r + ((lane & 16) ? 8 : 0);
        const float x = acc[mi][ni][r] + b1v;
        const float g = 0.5f * x * (1.0f + erff(x * 0.70710678118654752f));
        h1[((size_t)k * NROWS + n) * FHID + f] = f32_to_bf16_rne(g);
      }
    }
  }
}

// ---------------------------------------------------------------------------
// Kernel 3: out[n,d] = sum_k w[n,k] * (h1[k,n,:] @ W2[k,d,:] + b2[k,d])
// Grid (N/128, D/128); experts looped in-block, combine in registers.
// A tile (already bf16) staged with global_load_async_to_lds_b128 (ASYNCcnt);
// B tile staged through registers with f32->bf16 conversion.
// Single shared-memory block; buffer pointers computed arithmetically at each
// use (no LDS-pointer aggregates -> no addrspacecast in static initializers).
// ---------------------------------------------------------------------------
#define SM2_A0 0
#define SM2_A1 8192
#define SM2_B0 16384
#define SM2_B1 24576
#define SM2_W  32768
#define SM2_BYTES (SM2_W + 128 * KEXP * 4)

// Issue 128x32 bf16 A-tile as 512 async 16B transfers (2 per thread).
__device__ __forceinline__ void stage_A_async(const unsigned short* __restrict__ src,
                                              unsigned int abase, int t) {
#pragma unroll
  for (int i = 0; i < 2; ++i) {
    int linear = t + 256 * i;  // 0..511
    int row = linear >> 2;     // 0..127
    int c8  = linear & 3;      // chunk of 8 bf16
    async_copy_b128(abase + (unsigned int)(row * 32 + c8 * 8) * 2,
                    src + (size_t)row * FHID + c8 * 8);
  }
}

__global__ __launch_bounds__(256) void moe_gemm2_combine(
    const unsigned short* __restrict__ h1, const float* __restrict__ W2,
    const float* __restrict__ b2, const float* __restrict__ weights,
    float* __restrict__ out) {
  __shared__ __align__(16) unsigned char smem2[SM2_BYTES];

  const int t = threadIdx.x;
  const int lane = t & 31;
  const int wave = t >> 5;
  const int waveM = wave >> 2;
  const int waveN = wave & 3;
  const int nTile = blockIdx.x * 128;
  const int dTile = blockIdx.y * 128;

  float* ldsW = (float*)(smem2 + SM2_W);
#pragma unroll
  for (int i = 0; i < 4; ++i)
    ldsW[t + 256 * i] = weights[(size_t)nTile * KEXP + t + 256 * i];

  f32x8 tot[4][2];
#pragma unroll
  for (int mi = 0; mi < 4; ++mi)
#pragma unroll
    for (int ni = 0; ni < 2; ++ni)
#pragma unroll
      for (int r = 0; r < 8; ++r) tot[mi][ni][r] = 0.f;

  const int KT = FHID / 32;  // 32 stages per expert

  for (int k = 0; k < KEXP; ++k) {
    const unsigned short* __restrict__ h1k = h1 + ((size_t)k * NROWS + nTile) * FHID;
    const float* __restrict__ W2k = W2 + (size_t)k * DDIM * FHID + (size_t)dTile * FHID;

    f32x8 acc[4][2];
#pragma unroll
    for (int mi = 0; mi < 4; ++mi)
#pragma unroll
      for (int ni = 0; ni < 2; ++ni)
#pragma unroll
        for (int r = 0; r < 8; ++r) acc[mi][ni][r] = 0.f;

    float4 rb[4];
    // prologue: A goes straight to LDS via async DMA; B through regs+convert.
    stage_A_async(h1k, SM2_A0, t);
    tile_load_f32_T<128>(W2k, FHID, t, rb);
    tile_store_bf16_T<128>((unsigned short*)(smem2 + SM2_B0), t, rb);
    wait_async_zero();
    __syncthreads();

    for (int kk = 0; kk < KT; ++kk) {
      const int buf = kk & 1;
      const unsigned short* ldsAbuf =
          (const unsigned short*)(smem2 + SM2_A0 + (buf << 13));
      const unsigned short* ldsBbuf =
          (const unsigned short*)(smem2 + SM2_B0 + (buf << 13));

      if (kk + 1 < KT) {
        // safe: all readers of buf^1 finished before the previous barrier
        stage_A_async(h1k + (kk + 1) * 32,
                      (unsigned int)(SM2_A0 + ((buf ^ 1) << 13)), t);
        tile_load_f32_T<128>(W2k + (kk + 1) * 32, FHID, t, rb);
        if (kk + 2 < KT)
          __builtin_prefetch(W2k + (size_t)(t >> 1) * FHID + (kk + 2) * 32, 0, 3);
      }

      FragAB a[4], b[2];
#pragma unroll
      for (int mi = 0; mi < 4; ++mi)
        load_fragA(ldsAbuf, waveM * 64 + mi * 16, lane, a[mi]);
#pragma unroll
      for (int ni = 0; ni < 2; ++ni)
        load_fragB(ldsBbuf, waveN * 32 + ni * 16, lane, b[ni]);
#pragma unroll
      for (int mi = 0; mi < 4; ++mi)
#pragma unroll
        for (int ni = 0; ni < 2; ++ni)
          acc[mi][ni] = __builtin_amdgcn_wmma_f32_16x16x32_bf16(
              false, a[mi].v, false, b[ni].v, (short)0, acc[mi][ni], false, false);

      if (kk + 1 < KT) {
        tile_store_bf16_T<128>(
            (unsigned short*)(smem2 + SM2_B0 + ((buf ^ 1) << 13)), t, rb);
        wait_async_zero();  // our async A-transfers for buf^1 have landed
        __syncthreads();    // publish buf^1 to the workgroup
      }
    }

    // softmax-weighted combine (includes bias): tot += w[n,k]*(acc + b2[k,d])
#pragma unroll
    for (int ni = 0; ni < 2; ++ni) {
      const int d = dTile + waveN * 32 + ni * 16 + (lane & 15);
      const float b2v = b2[k * DDIM + d];
#pragma unroll
      for (int mi = 0; mi < 4; ++mi) {
#pragma unroll
        for (int r = 0; r < 8; ++r) {
          const int nl = waveM * 64 + mi * 16 + r + ((lane & 16) ? 8 : 0);
          const float w = ldsW[nl * KEXP + k];
          tot[mi][ni][r] += w * (acc[mi][ni][r] + b2v);
        }
      }
    }
  }

#pragma unroll
  for (int mi = 0; mi < 4; ++mi) {
#pragma unroll
    for (int ni = 0; ni < 2; ++ni) {
      const int d = dTile + waveN * 32 + ni * 16 + (lane & 15);
#pragma unroll
      for (int r = 0; r < 8; ++r) {
        const int n = nTile + waveM * 64 + mi * 16 + r + ((lane & 16) ? 8 : 0);
        out[(size_t)n * DDIM + d] = tot[mi][ni][r];
      }
    }
  }
}

// ---------------------------------------------------------------------------
extern "C" void kernel_launch(void* const* d_in, const int* in_sizes, int n_in,
                              void* d_out, int out_size, void* d_ws, size_t ws_size,
                              hipStream_t stream) {
  (void)in_sizes; (void)n_in; (void)out_size; (void)ws_size;

  const float* h_anchor = (const float*)d_in[0];
  const float* h_mask   = (const float*)d_in[1];
  const float* Wr       = (const float*)d_in[2];
  const float* br       = (const float*)d_in[3];
  const float* W1       = (const float*)d_in[4];
  const float* b1       = (const float*)d_in[5];
  const float* W2       = (const float*)d_in[6];
  const float* b2       = (const float*)d_in[7];
  float* out = (float*)d_out;

  // workspace: weights f32 [N,K] then h1 bf16 [K,N,F]  (~134.5 MB total)
  float* weights = (float*)d_ws;
  unsigned short* h1 = (unsigned short*)((char*)d_ws + WS_WEIGHTS_BYTES);

  moe_routing<<<NROWS / 8, 256, 0, stream>>>(h_mask, Wr, br, weights);
  moe_gemm1_gelu<<<dim3(NROWS / 128, FHID / 256, KEXP), 256, 0, stream>>>(
      h_anchor, h_mask, W1, b1, h1);
  moe_gemm2_combine<<<dim3(NROWS / 128, DDIM / 128), 256, 0, stream>>>(
      h1, W2, b2, weights, out);
}
